// CTCModel_33646773797486
// MI455X (gfx1250) — compile-verified, matrix operand order checked by reading, not run on hardware
//
#include <hip/hip_runtime.h>

#define DEVINL __device__ __forceinline__

typedef unsigned short u16;
typedef __attribute__((ext_vector_type(8)))  u16    v8u;
typedef __attribute__((ext_vector_type(16))) u16    v16u;
typedef __attribute__((ext_vector_type(16))) __bf16 v16bf;
typedef __attribute__((ext_vector_type(8)))  float  v8f;

// ---------------- helpers ----------------

DEVINL u16 f2bf(float x) {
  union { float f; unsigned u; } c; c.f = x;
  unsigned u = c.u;
  u = (u + 0x7FFFu + ((u >> 16) & 1u)) >> 16;   // RNE
  return (u16)u;
}

// 16-bit A/B fragment load per CDNA5 ISA layout:
// lanes 0-15 hold K={k0..k0+7} (elems 0-7) and K={k0+16..k0+23} (elems 8-15);
// lanes 16-31 hold K offsets +8.  Row index supplied by caller.
DEVINL v16bf load_frag(const u16* __restrict__ row, int k0, int lane) {
  int off = k0 + ((lane & 16) ? 8 : 0);
  v8u lo = *(const v8u*)(row + off);
  v8u hi = *(const v8u*)(row + off + 16);
  v16u v = __builtin_shufflevector(lo, hi, 0,1,2,3,4,5,6,7,8,9,10,11,12,13,14,15);
  return __builtin_bit_cast(v16bf, v);
}

DEVINL v8f wmma_bf16(v16bf a, v16bf b, v8f c) {
  return __builtin_amdgcn_wmma_f32_16x16x32_bf16(false, a, false, b, (short)0, c,
                                                 false, false);
}

DEVINL float sigmoidf_(float x) { return 1.f / (1.f + __expf(-x)); }

// ---------------- packing kernels ----------------

// X [rows x K] f32 -> Xp [rows x Kp] bf16, zero-padded K tail
__global__ void k_pad_x_bf16(const float* __restrict__ X, u16* __restrict__ Xp,
                             int rows, int K, int Kp) {
  long i = (long)blockIdx.x * blockDim.x + threadIdx.x;
  long total = (long)rows * Kp;
  if (i >= total) return;
  int r = (int)(i / Kp), c = (int)(i - (long)r * Kp);
  float v = (c < K) ? X[(long)r * K + c] : 0.f;
  Xp[i] = f2bf(v);
}

// Wt[n][kp] = W[(row0+kp)*ldw + col0 + n]  (transpose + bf16 + pad)
__global__ void k_pack_wt(const float* __restrict__ W, int ldw, int row0, int K,
                          int col0, int N, int Kp, int Np, u16* __restrict__ Wt) {
  long i = (long)blockIdx.x * blockDim.x + threadIdx.x;
  long total = (long)Np * Kp;
  if (i >= total) return;
  int n = (int)(i / Kp), kp = (int)(i - (long)n * Kp);
  float v = (n < N && kp < K) ? W[(long)(row0 + kp) * ldw + (col0 + n)] : 0.f;
  Wt[i] = f2bf(v);
}

// ---------------- generic WMMA GEMM ----------------
// C[M x N] = act(A[M x K](bf16) * Bt[N x K]^T(bf16) + bias)
// one wave per block; wave computes a 16(M) x 64(N) tile.
__global__ void __launch_bounds__(32) k_gemm_wmma(
    const u16* __restrict__ A, int lda,
    const u16* __restrict__ Bt, int ldb,
    const float* __restrict__ bias,
    float* __restrict__ outF, int ldoF,
    u16* __restrict__ outB, int ldoB, int obOff,
    int K, int Nstore, int act) {
  int lane = threadIdx.x & 31;
  int m0 = blockIdx.x * 16;
  int n0 = blockIdx.y * 64;
  const u16* arow = A + (size_t)(m0 + (lane & 15)) * lda;
  const u16* brow = Bt + (size_t)(n0 + (lane & 15)) * ldb;
  v8f acc[4] = {};
  for (int k0 = 0; k0 < K; k0 += 32) {
    v16bf a  = load_frag(arow, k0, lane);
    v16bf b0 = load_frag(brow,            k0, lane);
    v16bf b1 = load_frag(brow + 16 * (size_t)ldb, k0, lane);
    v16bf b2 = load_frag(brow + 32 * (size_t)ldb, k0, lane);
    v16bf b3 = load_frag(brow + 48 * (size_t)ldb, k0, lane);
    acc[0] = wmma_bf16(a, b0, acc[0]);
    acc[1] = wmma_bf16(a, b1, acc[1]);
    acc[2] = wmma_bf16(a, b2, acc[2]);
    acc[3] = wmma_bf16(a, b3, acc[3]);
  }
  int mbase = m0 + ((lane & 16) ? 8 : 0);
#pragma unroll
  for (int j = 0; j < 4; ++j) {
    int ncol = n0 + j * 16 + (lane & 15);
    if (ncol >= Nstore) continue;
    float bv = bias ? bias[ncol] : 0.f;
#pragma unroll
    for (int r = 0; r < 8; ++r) {
      int m = mbase + r;
      float v = acc[j][r] + bv;
      if (act) v = fmaxf(v, 0.f);
      if (outF) outF[(size_t)m * ldoF + ncol] = v;
      if (outB) outB[(size_t)m * ldoB + obOff + ncol] = f2bf(v);
    }
  }
}

// ---------------- persistent bidirectional GRU ----------------
// grid.x = 2 (direction); 1024 threads = 32 waves; h lives in LDS.
// Gx = x@Wgx + bg  (precomputed, natural time order), Cx = x@Wcx + bc.
// Per step:  gates = sigmoid(Gx[t_in] + h @ WghT)   -> r, u
//            c = tanh(Cx[t_in] + (r*h) @ WchT)
//            hn = u*h + (1-u)*c ; masked by seq_lens (dynamic_rnn semantics)
__global__ void __launch_bounds__(1024) k_gru(
    const float* __restrict__ Gx_f, const float* __restrict__ Cx_f,
    const u16* __restrict__ Wgh_f, const u16* __restrict__ Wch_f,
    const float* __restrict__ Gx_b, const float* __restrict__ Cx_b,
    const u16* __restrict__ Wgh_b, const u16* __restrict__ Wch_b,
    const int* __restrict__ seq_lens, u16* __restrict__ scores) {
  constexpr int Bn = 32, T = 1000, H = 512;
  extern __shared__ char smem[];
  float* hS  = (float*)smem;            // [32][512] f32 master state
  float* uS  = hS + Bn * H;             // [32][512] update gate
  u16*   hB  = (u16*)(uS + Bn * H);     // [32][512] bf16 state (WMMA A)
  u16*   rhB = hB + Bn * H;             // [32][512] bf16 r*h   (WMMA A)
  int*   lenS = (int*)(rhB + Bn * H);   // [32]

  int dir = blockIdx.x;
  const float* Gx = dir ? Gx_b : Gx_f;
  const float* Cx = dir ? Cx_b : Cx_f;
  const u16* Wgh = dir ? Wgh_b : Wgh_f;
  const u16* Wch = dir ? Wch_b : Wch_f;

  int tid = threadIdx.x;
  int lane = tid & 31, w = tid >> 5;
  for (int i = tid; i < Bn * H; i += 1024) { hS[i] = 0.f; hB[i] = 0; }
  if (tid < Bn) lenS[tid] = seq_lens[tid];
  __syncthreads();

  int mt = w & 1;                 // which 16-row batch tile
  int m0 = mt * 16;
  int ng = w >> 1;                // 0..15
  int gn0 = ng * 64;              // gate N base (4 tiles of 16)
  int cn0 = ng * 32;              // candidate N base (2 tiles of 16)
  const u16* arowG = hB  + (size_t)(m0 + (lane & 15)) * H;
  const u16* arowC = rhB + (size_t)(m0 + (lane & 15)) * H;
  const u16* browG = Wgh + (size_t)(gn0 + (lane & 15)) * H;
  const u16* browC = Wch + (size_t)(cn0 + (lane & 15)) * H;
  int mbase = m0 + ((lane & 16) ? 8 : 0);

  for (int t = 0; t < T; ++t) {
    // ---- Phase A: gates  G[32x1024] = hB @ WghT ----
    {
      v8f acc[4] = {};
#pragma unroll 4
      for (int k0 = 0; k0 < H; k0 += 32) {
        v16bf a  = load_frag(arowG, k0, lane);
        v16bf b0 = load_frag(browG,                 k0, lane);
        v16bf b1 = load_frag(browG + 16 * (size_t)H, k0, lane);
        v16bf b2 = load_frag(browG + 32 * (size_t)H, k0, lane);
        v16bf b3 = load_frag(browG + 48 * (size_t)H, k0, lane);
        acc[0] = wmma_bf16(a, b0, acc[0]);
        acc[1] = wmma_bf16(a, b1, acc[1]);
        acc[2] = wmma_bf16(a, b2, acc[2]);
        acc[3] = wmma_bf16(a, b3, acc[3]);
      }
#pragma unroll
      for (int j = 0; j < 4; ++j) {
        int ncol = gn0 + j * 16 + (lane & 15);
#pragma unroll
        for (int r = 0; r < 8; ++r) {
          int m = mbase + r;
          int len = lenS[m];
          int tin = dir ? ((t < len) ? len - 1 - t : t) : t;
          float g = acc[j][r] + Gx[((size_t)m * T + tin) * 1024 + ncol];
          float s = sigmoidf_(g);
          if (ncol < H) rhB[m * H + ncol] = f2bf(s * hS[m * H + ncol]);   // r-gate
          else          uS[m * H + (ncol - H)] = s;                        // u-gate
        }
      }
    }
    __syncthreads();
    // ---- Phase B: candidate + state update ----
    {
      v8f acc[2] = {};
#pragma unroll 4
      for (int k0 = 0; k0 < H; k0 += 32) {
        v16bf a  = load_frag(arowC, k0, lane);
        v16bf b0 = load_frag(browC,                 k0, lane);
        v16bf b1 = load_frag(browC + 16 * (size_t)H, k0, lane);
        acc[0] = wmma_bf16(a, b0, acc[0]);
        acc[1] = wmma_bf16(a, b1, acc[1]);
      }
#pragma unroll
      for (int j = 0; j < 2; ++j) {
        int ncol = cn0 + j * 16 + (lane & 15);
#pragma unroll
        for (int r = 0; r < 8; ++r) {
          int m = mbase + r;
          int len = lenS[m];
          int tin = dir ? ((t < len) ? len - 1 - t : t) : t;
          float cin = acc[j][r] + Cx[((size_t)m * T + tin) * H + ncol];
          float c = tanhf(cin);
          float u = uS[m * H + ncol];
          float hold = hS[m * H + ncol];
          float hn = u * hold + (1.f - u) * c;
          bool mask = t < len;
          float hnew = mask ? hn : hold;
          hS[m * H + ncol] = hnew;
          hB[m * H + ncol] = f2bf(hnew);
          // output goes to natural time position; masked positions -> 0
          scores[((size_t)m * T + tin) * 1024 + dir * H + ncol] =
              mask ? f2bf(hn) : (u16)0;
        }
      }
    }
    __syncthreads();
  }
}

// ---------------- launch ----------------

extern "C" void kernel_launch(void* const* d_in, const int* in_sizes, int n_in,
                              void* d_out, int out_size, void* d_ws, size_t ws_size,
                              hipStream_t stream) {
  (void)in_sizes; (void)n_in; (void)out_size; (void)ws_size;
  constexpr int Bn = 32, T = 1000, F = 161, H = 512, C = 62;
  constexpr int M = Bn * T;          // 32000
  constexpr int Kx = 192;            // 161 padded to mult of 32

  const float* x    = (const float*)d_in[0];
  const int*   lens = (const int*)  d_in[1];
  const float* w1   = (const float*)d_in[2];
  const float* b1   = (const float*)d_in[3];
  const float* w2   = (const float*)d_in[4];
  const float* b2   = (const float*)d_in[5];
  const float* wg_f = (const float*)d_in[6];
  const float* bg_f = (const float*)d_in[7];
  const float* wc_f = (const float*)d_in[8];
  const float* bc_f = (const float*)d_in[9];
  const float* wg_b = (const float*)d_in[10];
  const float* bg_b = (const float*)d_in[11];
  const float* wc_b = (const float*)d_in[12];
  const float* bc_b = (const float*)d_in[13];
  const float* wf   = (const float*)d_in[14];
  const float* bf   = (const float*)d_in[15];

  char* ws = (char*)d_ws;
  size_t off = 0;
  auto alloc = [&](size_t bytes) -> char* {
    char* p = ws + off;
    off = (off + bytes + 255) & ~(size_t)255;
    return p;
  };

  u16* Xpad  = (u16*)alloc((size_t)M * Kx * 2);
  u16* W1t   = (u16*)alloc((size_t)H * Kx * 2);
  u16* W2t   = (u16*)alloc((size_t)H * H * 2);
  u16* Wgx_f = (u16*)alloc((size_t)1024 * H * 2);
  u16* Wgh_f = (u16*)alloc((size_t)1024 * H * 2);
  u16* Wcx_f = (u16*)alloc((size_t)H * H * 2);
  u16* Wch_f = (u16*)alloc((size_t)H * H * 2);
  u16* Wgx_b = (u16*)alloc((size_t)1024 * H * 2);
  u16* Wgh_b = (u16*)alloc((size_t)1024 * H * 2);
  u16* Wcx_b = (u16*)alloc((size_t)H * H * 2);
  u16* Wch_b = (u16*)alloc((size_t)H * H * 2);
  u16* Wft   = (u16*)alloc((size_t)64 * 1024 * 2);
  u16* H1    = (u16*)alloc((size_t)M * H * 2);
  u16* H2    = (u16*)alloc((size_t)M * H * 2);
  float* GxF = (float*)alloc((size_t)M * 1024 * 4);
  float* GxB = (float*)alloc((size_t)M * 1024 * 4);
  float* CxF = (float*)alloc((size_t)M * H * 4);
  float* CxB = (float*)alloc((size_t)M * H * 4);
  u16* scores = (u16*)alloc((size_t)M * 1024 * 2);

  auto cdiv = [](long a, long b) { return (int)((a + b - 1) / b); };

  // pack inputs / weights to bf16
  k_pad_x_bf16<<<cdiv((long)M * Kx, 256), 256, 0, stream>>>(x, Xpad, M, F, Kx);
  k_pack_wt<<<cdiv((long)H * Kx, 256),     256, 0, stream>>>(w1,   H,    0,   F,    0, H,    Kx, H,    W1t);
  k_pack_wt<<<cdiv((long)H * H, 256),      256, 0, stream>>>(w2,   H,    0,   H,    0, H,    H,  H,    W2t);
  k_pack_wt<<<cdiv((long)1024 * H, 256),   256, 0, stream>>>(wg_f, 1024, 0,   H,    0, 1024, H,  1024, Wgx_f);
  k_pack_wt<<<cdiv((long)1024 * H, 256),   256, 0, stream>>>(wg_f, 1024, H,   H,    0, 1024, H,  1024, Wgh_f);
  k_pack_wt<<<cdiv((long)H * H, 256),      256, 0, stream>>>(wc_f, H,    0,   H,    0, H,    H,  H,    Wcx_f);
  k_pack_wt<<<cdiv((long)H * H, 256),      256, 0, stream>>>(wc_f, H,    H,   H,    0, H,    H,  H,    Wch_f);
  k_pack_wt<<<cdiv((long)1024 * H, 256),   256, 0, stream>>>(wg_b, 1024, 0,   H,    0, 1024, H,  1024, Wgx_b);
  k_pack_wt<<<cdiv((long)1024 * H, 256),   256, 0, stream>>>(wg_b, 1024, H,   H,    0, 1024, H,  1024, Wgh_b);
  k_pack_wt<<<cdiv((long)H * H, 256),      256, 0, stream>>>(wc_b, H,    0,   H,    0, H,    H,  H,    Wcx_b);
  k_pack_wt<<<cdiv((long)H * H, 256),      256, 0, stream>>>(wc_b, H,    H,   H,    0, H,    H,  H,    Wch_b);
  k_pack_wt<<<cdiv((long)64 * 1024, 256),  256, 0, stream>>>(wf,   C,    0,   1024, 0, C,    1024, 64, Wft);

  // feed-forward layers (relu fused), bf16 outputs
  k_gemm_wmma<<<dim3(M / 16, H / 64), 32, 0, stream>>>(
      Xpad, Kx, W1t, Kx, b1, nullptr, 0, H1, H, 0, Kx, H, 1);
  k_gemm_wmma<<<dim3(M / 16, H / 64), 32, 0, stream>>>(
      H1, H, W2t, H, b2, nullptr, 0, H2, H, 0, H, H, 1);

  // time-parallel GRU input contributions (bias folded in), f32 outputs
  k_gemm_wmma<<<dim3(M / 16, 1024 / 64), 32, 0, stream>>>(
      H2, H, Wgx_f, H, bg_f, GxF, 1024, nullptr, 0, 0, H, 1024, 0);
  k_gemm_wmma<<<dim3(M / 16, 1024 / 64), 32, 0, stream>>>(
      H2, H, Wgx_b, H, bg_b, GxB, 1024, nullptr, 0, 0, H, 1024, 0);
  k_gemm_wmma<<<dim3(M / 16, H / 64), 32, 0, stream>>>(
      H2, H, Wcx_f, H, bc_f, CxF, H, nullptr, 0, 0, H, H, 0);
  k_gemm_wmma<<<dim3(M / 16, H / 64), 32, 0, stream>>>(
      H2, H, Wcx_b, H, bc_b, CxB, H, nullptr, 0, 0, H, H, 0);

  // persistent recurrence, both directions concurrently
  size_t smem = (size_t)(2 * Bn * H) * 4 + (size_t)(2 * Bn * H) * 2 + Bn * 4;
  (void)hipFuncSetAttribute((const void*)k_gru,
                            hipFuncAttributeMaxDynamicSharedMemorySize, (int)smem);
  k_gru<<<2, 1024, smem, stream>>>(GxF, CxF, Wgh_f, Wch_f,
                                   GxB, CxB, Wgh_b, Wch_b,
                                   lens, scores);

  // final projection [M x 1024] x [1024 x 62]
  k_gemm_wmma<<<dim3(M / 16, 1), 32, 0, stream>>>(
      scores, 1024, Wft, 1024, bf, (float*)d_out, C, nullptr, 0, 0, 1024, C, 0);
}
